// CharBiLSTM_50912542327067
// MI455X (gfx1250) — matrix operand order repaired and checked by reading, not compiled, not run on hardware
//
#include <hip/hip_runtime.h>

typedef __attribute__((ext_vector_type(16))) _Float16 v16h;
typedef __attribute__((ext_vector_type(8)))  float    v8f;

#define NWORD 4096
#define LL 16
#define EE 128
#define HH 512
#define MT 32            // rows (words) per block

__device__ __forceinline__ float sigmoidf_(float x) { return 1.0f / (1.0f + __expf(-x)); }

__device__ __forceinline__ v8f wmma16(v16h a, v16h b, v8f c) {
  return __builtin_amdgcn_wmma_f32_16x16x32_f16(false, a, false, b, (short)0, c, false, false);
}

union AF { v16h h; unsigned int u[8]; };

// A fragment (16x32 f16) from an LDS tile, row-major, row stride `stride` halves,
// K-tile at column k0.  lanes0-15: rows 0-15, K{0..7,16..23}; lanes16-31: K{8..15,24..31}.
__device__ __forceinline__ v16h ldsAfrag(const _Float16* base, int stride, int k0, int lane) {
  AF f;
  const _Float16* p = base + (lane & 15) * stride + k0 + ((lane >> 4) << 3);
#pragma unroll
  for (int r = 0; r < 8; ++r) {
    int k = (r < 4) ? (2 * r) : (2 * r + 8);
    f.u[r] = *(const unsigned int*)(p + k);
  }
  return f.h;
}

// B fragment (32x16 f16) from weight stored n-major W[n][k], leading dim ldk.
// lanes0-15: K{0..7,16..23}, lanes16-31: K{8..15,24..31}, col = lane%16; (k,k+1) contiguous.
__device__ __forceinline__ v16h gblBfrag(const _Float16* __restrict__ W, int n0, int ldk,
                                         int k0, int lane) {
  AF f;
  const _Float16* p = W + (size_t)(n0 + (lane & 15)) * ldk + k0 + ((lane >> 4) << 3);
#pragma unroll
  for (int r = 0; r < 8; ++r) f.u[r] = *(const unsigned int*)(p + ((r < 4) ? (2 * r) : (2 * r + 8)));
  return f.h;
}

// One K-step of the 4-gate x 2-rowtile gate GEMM: all loads first (distinct live
// regs -> partial loadcnt waits), then an 8-deep WMMA burst.  Each B frag feeds 2 WMMAs.
__device__ __forceinline__ void gemm_step(const _Float16* abase, int astride,
                                          const _Float16* __restrict__ W, int ldk,
                                          int jcol, int k0, int lane, v8f acc[4][2]) {
  v16h a0 = ldsAfrag(abase, astride, k0, lane);
  v16h a1 = ldsAfrag(abase + 16 * astride, astride, k0, lane);
  v16h b0 = gblBfrag(W, 0 * HH + jcol, ldk, k0, lane);
  v16h b1 = gblBfrag(W, 1 * HH + jcol, ldk, k0, lane);
  v16h b2 = gblBfrag(W, 2 * HH + jcol, ldk, k0, lane);
  v16h b3 = gblBfrag(W, 3 * HH + jcol, ldk, k0, lane);
  acc[0][0] = wmma16(a0, b0, acc[0][0]);
  acc[0][1] = wmma16(a1, b0, acc[0][1]);
  acc[1][0] = wmma16(a0, b1, acc[1][0]);
  acc[1][1] = wmma16(a1, b1, acc[1][1]);
  acc[2][0] = wmma16(a0, b2, acc[2][0]);
  acc[2][1] = wmma16(a1, b2, acc[2][1]);
  acc[3][0] = wmma16(a0, b3, acc[3][0]);
  acc[3][1] = wmma16(a1, b3, acc[3][1]);
}

// Single prologue kernel: all f32->f16 weight/embedding converts + bias folds.
#define SZ_EMB   (256 * 128)
#define SZ_WIH   (2048 * 128)
#define SZ_WHH   (2048 * 512)
#define SZ_WP    (512 * 1024)
#define SZ_B     2048
#define PREP_TOT (SZ_EMB + 2 * (SZ_WIH + SZ_WHH) + SZ_WP + 2 * SZ_B)

__global__ void prep(const float* __restrict__ emb,
                     const float* __restrict__ WihF, const float* __restrict__ WhhF,
                     const float* __restrict__ WihB, const float* __restrict__ WhhB,
                     const float* __restrict__ Wp,
                     const float* __restrict__ bihF, const float* __restrict__ bhhF,
                     const float* __restrict__ bihB, const float* __restrict__ bhhB,
                     _Float16* __restrict__ demb,
                     _Float16* __restrict__ dWihF, _Float16* __restrict__ dWhhF,
                     _Float16* __restrict__ dWihB, _Float16* __restrict__ dWhhB,
                     _Float16* __restrict__ dWp,
                     float* __restrict__ dbF, float* __restrict__ dbB) {
  long o = (long)blockIdx.x * blockDim.x + threadIdx.x;
  if (o < SZ_EMB) { demb[o]  = (_Float16)emb[o];  return; } o -= SZ_EMB;
  if (o < SZ_WIH) { dWihF[o] = (_Float16)WihF[o]; return; } o -= SZ_WIH;
  if (o < SZ_WHH) { dWhhF[o] = (_Float16)WhhF[o]; return; } o -= SZ_WHH;
  if (o < SZ_WIH) { dWihB[o] = (_Float16)WihB[o]; return; } o -= SZ_WIH;
  if (o < SZ_WHH) { dWhhB[o] = (_Float16)WhhB[o]; return; } o -= SZ_WHH;
  if (o < SZ_WP)  { dWp[o]   = (_Float16)Wp[o];   return; } o -= SZ_WP;
  if (o < SZ_B)   { dbF[o]   = bihF[o] + bhhF[o]; return; } o -= SZ_B;
  if (o < SZ_B)   { dbB[o]   = bihB[o] + bhhB[o]; return; }
}

// One direction's LSTM over a 32-word tile; compile-time OUTOFF/BWD so addressing
// folds to constants.  LDS buffers are shared between both template instantiations.
template<int OUTOFF, bool BWD>
__device__ __forceinline__
void lstm_body(_Float16 (*xbuf)[MT][EE], _Float16 (*hbuf)[MT][HH], int* llen,
               const int* __restrict__ chars, const int* __restrict__ lens,
               const _Float16* __restrict__ emb16,
               const _Float16* __restrict__ Wih,   // [2048][128] n-major
               const _Float16* __restrict__ Whh,   // [2048][512] n-major
               const float* __restrict__ bias,     // [2048] = bih+bhh
               _Float16* __restrict__ cat,         // [4096][1024]
               int blk)
{
  const int tid  = threadIdx.x;
  const int lane = tid & 31;
  const int wave = tid >> 5;               // 0..7
  const int row0 = blk * MT;

  if (tid < MT) llen[tid] = lens[row0 + tid];
  {
    unsigned int* hz = (unsigned int*)&hbuf[0][0][0];
    for (int i = tid; i < MT * HH / 2; i += 256) hz[i] = 0u;  // h0 = 0
  }
  __syncthreads();

  const int xr = tid >> 4;            // rows xr and xr+16 gathered by this thread
  const int xc = (tid & 15) * 8;      // 8 halves = 16B per row chunk
  {
    int l0 = llen[xr], l1 = llen[xr + 16];
    int c0 = BWD ? (l0 - 1) : 0; if (c0 < 0) c0 = 0;
    int c1 = BWD ? (l1 - 1) : 0; if (c1 < 0) c1 = 0;
    int id0 = chars[(row0 + xr) * LL + c0];
    int id1 = chars[(row0 + xr + 16) * LL + c1];
    *(uint4*)&xbuf[0][xr][xc]      = *(const uint4*)(emb16 + id0 * EE + xc);
    *(uint4*)&xbuf[0][xr + 16][xc] = *(const uint4*)(emb16 + id1 * EE + xc);
  }
  __syncthreads();

  const int jbase = wave * 64;        // this wave's 64 hidden columns
  const int colN  = lane & 15;
  const int rbase = (lane >> 4) << 3; // C/D rows v / v+8

  v8f cst[4][2];                      // cell state, persistent across steps
#pragma unroll
  for (int jt = 0; jt < 4; ++jt)
#pragma unroll
    for (int rt = 0; rt < 2; ++rt)
#pragma unroll
      for (int e = 0; e < 8; ++e) cst[jt][rt][e] = 0.0f;

  for (int t = 0; t < LL; ++t) {
    const int cur = t & 1, nxt = cur ^ 1;

    // prefetch next step's char embeddings into the other x buffer
    if (t + 1 < LL) {
      int l0 = llen[xr], l1 = llen[xr + 16];
      int c0 = BWD ? (l0 - 2 - t) : (t + 1); if (c0 < 0) c0 = 0;
      int c1 = BWD ? (l1 - 2 - t) : (t + 1); if (c1 < 0) c1 = 0;
      int id0 = chars[(row0 + xr) * LL + c0];
      int id1 = chars[(row0 + xr + 16) * LL + c1];
      *(uint4*)&xbuf[nxt][xr][xc]      = *(const uint4*)(emb16 + id0 * EE + xc);
      *(uint4*)&xbuf[nxt][xr + 16][xc] = *(const uint4*)(emb16 + id1 * EE + xc);
    }

#pragma unroll
    for (int jt = 0; jt < 4; ++jt) {
      const int jcol = jbase + jt * 16;
      v8f acc[4][2];                            // [gate][rowtile]
#pragma unroll
      for (int g = 0; g < 4; ++g) {
        float bv = bias[g * HH + jcol + colN];  // same column in all 8 rows
#pragma unroll
        for (int rt = 0; rt < 2; ++rt)
#pragma unroll
          for (int e = 0; e < 8; ++e) acc[g][rt][e] = bv;
      }
      // input projection part: K = 128
#pragma unroll
      for (int kc = 0; kc < 4; ++kc)
        gemm_step(&xbuf[cur][0][0], EE, Wih, EE, jcol, kc * 32, lane, acc);
      // recurrent part: K = 512
#pragma unroll 2
      for (int kc = 0; kc < 16; ++kc)
        gemm_step(&hbuf[cur][0][0], HH, Whh, HH, jcol, kc * 32, lane, acc);

      // elementwise LSTM cell update (gate order i,f,g,o)
#pragma unroll
      for (int rt = 0; rt < 2; ++rt) {
#pragma unroll
        for (int e = 0; e < 8; ++e) {
          int r = rt * 16 + e + rbase;
          float iv = sigmoidf_(acc[0][rt][e]);
          float fv = sigmoidf_(acc[1][rt][e]);
          float gv = tanhf(acc[2][rt][e]);
          float ov = sigmoidf_(acc[3][rt][e]);
          float c  = fv * cst[jt][rt][e] + iv * gv;
          cst[jt][rt][e] = c;
          float h  = ov * tanhf(c);
          hbuf[nxt][r][jcol + colN] = (_Float16)h;
          if (t == llen[r] - 1)
            cat[(size_t)(row0 + r) * 1024 + OUTOFF + jcol + colN] = (_Float16)h;
        }
      }
      // scheduling fence: keep the register window per j-tile (prevents the
      // allocator from hoisting the next tile's 120 fragment loads over this one)
      asm volatile("" ::: "memory");
    }
    __syncthreads();
  }
}

// Both directions in one launch: even blocks = forward, odd blocks = backward.
__global__ __launch_bounds__(256)
void lstm_both(const int* __restrict__ chars, const int* __restrict__ lens,
               const _Float16* __restrict__ emb16,
               const _Float16* __restrict__ WihF, const _Float16* __restrict__ WhhF,
               const float* __restrict__ biasF,
               const _Float16* __restrict__ WihB, const _Float16* __restrict__ WhhB,
               const float* __restrict__ biasB,
               _Float16* __restrict__ cat)
{
  __shared__ _Float16 xbuf[2][MT][EE];
  __shared__ _Float16 hbuf[2][MT][HH];
  __shared__ int llen[MT];
  const int blk = blockIdx.x >> 1;
  if (blockIdx.x & 1)
    lstm_body<0, true >(xbuf, hbuf, llen, chars, lens, emb16, WihB, WhhB, biasB, cat, blk);
  else
    lstm_body<512, false>(xbuf, hbuf, llen, chars, lens, emb16, WihF, WhhF, biasF, cat, blk);
}

// out[4096][512] = concat[4096][1024] @ Wp^T + bp   (Wp given n-major [512][1024])
__global__ __launch_bounds__(256)
void proj_kernel(const _Float16* __restrict__ cat, const _Float16* __restrict__ Wp16,
                 const float* __restrict__ bp, float* __restrict__ out)
{
  const int tid  = threadIdx.x;
  const int lane = tid & 31;
  const int wave = tid >> 5;
  const int row0 = blockIdx.x * MT;
  const int jbase = wave * 64;
  const int colN  = lane & 15;
  const int rbase = (lane >> 4) << 3;

  v8f acc[4][2];
#pragma unroll
  for (int jt = 0; jt < 4; ++jt) {
    float bv = bp[jbase + jt * 16 + colN];
#pragma unroll
    for (int rt = 0; rt < 2; ++rt)
#pragma unroll
      for (int e = 0; e < 8; ++e) acc[jt][rt][e] = bv;
  }

  const int arow0 = row0 + (lane & 15);
  const int khi   = (lane >> 4) << 3;
#pragma unroll 2
  for (int kc = 0; kc < 32; ++kc) {          // K = 1024
    AF a0, a1;
    const _Float16* p0 = cat + (size_t)arow0 * 1024 + kc * 32 + khi;
    const _Float16* p1 = p0 + (size_t)16 * 1024;
#pragma unroll
    for (int r = 0; r < 8; ++r) {
      int k = (r < 4) ? (2 * r) : (2 * r + 8);
      a0.u[r] = *(const unsigned int*)(p0 + k);
      a1.u[r] = *(const unsigned int*)(p1 + k);
    }
    v16h b0 = gblBfrag(Wp16, jbase + 0 * 16, 1024, kc * 32, lane);
    v16h b1 = gblBfrag(Wp16, jbase + 1 * 16, 1024, kc * 32, lane);
    v16h b2 = gblBfrag(Wp16, jbase + 2 * 16, 1024, kc * 32, lane);
    v16h b3 = gblBfrag(Wp16, jbase + 3 * 16, 1024, kc * 32, lane);
    acc[0][0] = wmma16(a0.h, b0, acc[0][0]);
    acc[0][1] = wmma16(a1.h, b0, acc[0][1]);
    acc[1][0] = wmma16(a0.h, b1, acc[1][0]);
    acc[1][1] = wmma16(a1.h, b1, acc[1][1]);
    acc[2][0] = wmma16(a0.h, b2, acc[2][0]);
    acc[2][1] = wmma16(a1.h, b2, acc[2][1]);
    acc[3][0] = wmma16(a0.h, b3, acc[3][0]);
    acc[3][1] = wmma16(a1.h, b3, acc[3][1]);
  }
#pragma unroll
  for (int jt = 0; jt < 4; ++jt)
#pragma unroll
    for (int rt = 0; rt < 2; ++rt)
#pragma unroll
      for (int e = 0; e < 8; ++e)
        out[(size_t)(row0 + rt * 16 + e + rbase) * 512 + jbase + jt * 16 + colN] = acc[jt][rt][e];
}

extern "C" void kernel_launch(void* const* d_in, const int* in_sizes, int n_in,
                              void* d_out, int out_size, void* d_ws, size_t ws_size,
                              hipStream_t stream) {
  (void)in_sizes; (void)n_in; (void)out_size; (void)ws_size;
  const int*   chars = (const int*)d_in[0];
  const int*   lens  = (const int*)d_in[1];
  const float* emb   = (const float*)d_in[2];
  const float* Wih_f = (const float*)d_in[3];
  const float* Whh_f = (const float*)d_in[4];
  const float* bih_f = (const float*)d_in[5];
  const float* bhh_f = (const float*)d_in[6];
  const float* Wih_b = (const float*)d_in[7];
  const float* Whh_b = (const float*)d_in[8];
  const float* bih_b = (const float*)d_in[9];
  const float* bhh_b = (const float*)d_in[10];
  const float* Wp    = (const float*)d_in[11];
  const float* bp    = (const float*)d_in[12];
  float* out = (float*)d_out;

  char* ws = (char*)d_ws;
  size_t off = 0;
  auto take = [&](size_t bytes) -> void* {
    void* p = ws + off;
    off += (bytes + 255) & ~(size_t)255;
    return p;
  };
  _Float16* emb16  = (_Float16*)take((size_t)SZ_EMB * 2);
  _Float16* WihF16 = (_Float16*)take((size_t)SZ_WIH * 2);
  _Float16* WhhF16 = (_Float16*)take((size_t)SZ_WHH * 2);
  _Float16* WihB16 = (_Float16*)take((size_t)SZ_WIH * 2);
  _Float16* WhhB16 = (_Float16*)take((size_t)SZ_WHH * 2);
  _Float16* Wp16   = (_Float16*)take((size_t)SZ_WP * 2);
  float*    biasF  = (float*)take((size_t)SZ_B * 4);
  float*    biasB  = (float*)take((size_t)SZ_B * 4);
  _Float16* cat    = (_Float16*)take((size_t)NWORD * 1024 * 2);

  prep<<<(PREP_TOT + 255) / 256, 256, 0, stream>>>(
      emb, Wih_f, Whh_f, Wih_b, Whh_b, Wp, bih_f, bhh_f, bih_b, bhh_b,
      emb16, WihF16, WhhF16, WihB16, WhhB16, Wp16, biasF, biasB);

  // 256 blocks: even = forward (cat cols 512..1023), odd = backward (cols 0..511)
  lstm_both<<<2 * (NWORD / MT), 256, 0, stream>>>(chars, lens, emb16,
                                                  WihF16, WhhF16, biasF,
                                                  WihB16, WhhB16, biasB, cat);

  proj_kernel<<<NWORD / MT, 256, 0, stream>>>(cat, Wp16, bp, out);
}
